// RelativePositionEncoder_70093866271174
// MI455X (gfx1250) — matrix operand order repaired and checked by reading, not compiled, not run on hardware
//
#include <hip/hip_runtime.h>
#include <math.h>

// MI455X / gfx1250, wave32. WMMA f16->f32 for all GEMMs.
typedef __attribute__((ext_vector_type(16))) _Float16 v16h;
typedef __attribute__((ext_vector_type(8)))  float    v8f;

#define KAG  50
#define FDIM 6
#define HDIM 128
#define DDIM 256
#define NSC  256   // B*T scenes

// ---- WMMA fragment helpers (contiguous k order, same for A and B) ----

// A: 16x32 f16, row-major source with leading dim lda (in halves).
// lane L: row = m0 + (L&15); k = k0 + (L>=16?16:0) + i  (16 contiguous halves)
__device__ inline v16h load_a_frag(const _Float16* A, int lda, int m0, int k0, int lane) {
  const _Float16* p = A + (m0 + (lane & 15)) * lda + k0 + ((lane >> 4) << 4);
  v16h a;
#pragma unroll
  for (int g = 0; g < 2; ++g)
#pragma unroll
    for (int w = 0; w < 8; ++w)
      a[g * 8 + w] = p[g * 8 + w];
  return a;
}

// B: 32x16 f16 from N-major (transposed) storage Bt[n*ldb + k].
// lane L: col = n0 + (L&15); k = k0 + (L>=16?16:0) + i  (16 contiguous halves)
__device__ inline v16h load_bt_frag(const _Float16* Bt, int ldb, int n0, int k0, int lane) {
  const _Float16* p = Bt + (n0 + (lane & 15)) * ldb + k0 + ((lane >> 4) << 4);
  v16h b;
#pragma unroll
  for (int i = 0; i < 16; ++i) b[i] = p[i];
  return b;
}

__device__ inline v8f wmma16(v16h a, v16h b, v8f c) {
  return __builtin_amdgcn_wmma_f32_16x16x32_f16(false, a, false, b, (short)0, c, false, false);
}

// ---- kernel0: convert f32 [K,N] weight -> f16 transposed [N,K] ----
__global__ void convert_t_kernel(const float* __restrict__ src, _Float16* __restrict__ dst,
                                 int K, int N) {
  int idx = blockIdx.x * 256 + threadIdx.x;
  if (idx >= K * N) return;
  int k = idx / N, n = idx - k * N;
  dst[n * K + k] = (_Float16)src[k * N + n];
}

// ---- kernel1: fully fused per-scene pipeline (1 block = 1 scene, 256 thr = 8 waves) ----
__global__ void __launch_bounds__(256)
scene_kernel(const float* __restrict__ states, const float* __restrict__ masks,
             const float* __restrict__ w_a, const float* __restrict__ b_a,
             const float* __restrict__ w_r1, const float* __restrict__ b_r1,
             const float* __restrict__ b_r2, const float* __restrict__ b_g,
             const _Float16* __restrict__ wt_xi, const _Float16* __restrict__ wt_xj,
             const _Float16* __restrict__ wt_r2, const _Float16* __restrict__ wt_g1,
             const _Float16* __restrict__ wt_g2,
             _Float16* __restrict__ pooled_out) {
  __shared__ float    s_sh[KAG * FDIM];
  __shared__ float    m_sh[64];             // padded, rows 50..63 = 0
  __shared__ float4   geo_sh[2][64];        // double-buffered (dist, angle, sdiff, mask_j)
  __shared__ float    S_sh;
  __shared__ _Float16 af_h[64 * HDIM];      // agent_feat f16, pad rows zeroed
  __shared__ float    xi_sh[64 * HDIM];
  __shared__ float    xj_sh[64 * HDIM];     // later reused as hn_h (f16, 64 rows)
  __shared__ float    hsum_sh[KAG * HDIM];  // later reused as agg_h (f16, 64 rows)
  __shared__ float    pooled_sh[HDIM];

  const int tid = threadIdx.x;
  const int lane = tid & 31;
  const int wave = tid >> 5;
  const int sc = blockIdx.x;

  // Per-wave WMMA tile mapping: one 16-row band, four 16-col tiles.
  const int mt0 = (wave >> 1) << 4;          // 0,16,32,48
  const int nb0 = (wave & 1) << 6;           // 0 or 64
  const int nn_c = (lane & 15);              // col within tile
  const int rb_c = ((lane >> 4) << 3);       // row base within tile (0 or 8)

  // P0: stage inputs, zero accumulators
  for (int i = tid; i < KAG * FDIM; i += 256) s_sh[i] = states[sc * KAG * FDIM + i];
  if (tid < 64) m_sh[tid] = (tid < KAG) ? masks[sc * KAG + tid] : 0.f;
  for (int i = tid; i < KAG * HDIM; i += 256) hsum_sh[i] = 0.f;
  if (tid < HDIM) pooled_sh[tid] = 0.f;
  __syncthreads();

  if (tid == 0) { float s = 0.f; for (int i = 0; i < KAG; ++i) s += m_sh[i]; S_sh = s; }

  // P1: agent_feat = relu(s @ w_a + b_a) -> f16, pad rows zeroed
  {
    const int h = tid & 127;
    float wa[FDIM];
#pragma unroll
    for (int f = 0; f < FDIM; ++f) wa[f] = w_a[f * HDIM + h];
    const float ba = b_a[h];
    for (int i = tid >> 7; i < 64; i += 2) {
      float v = 0.f;
      if (i < KAG) {
        float acc = ba;
#pragma unroll
        for (int f = 0; f < FDIM; ++f) acc += s_sh[i * FDIM + f] * wa[f];
        v = fmaxf(acc, 0.f);
      }
      af_h[i * HDIM + h] = (_Float16)v;
    }
  }
  __syncthreads();

  // P2: x_i = af@w_r1[:H], x_j = af@w_r1[H:2H]. Share A frags across both GEMMs & 4 n-tiles.
  {
    v8f accI[4] = {}, accJ[4] = {};
#pragma unroll
    for (int kc = 0; kc < 4; ++kc) {
      const v16h a = load_a_frag(af_h, HDIM, mt0, kc * 32, lane);
#pragma unroll
      for (int c = 0; c < 4; ++c) {
        accI[c] = wmma16(a, load_bt_frag(wt_xi, HDIM, nb0 + c * 16, kc * 32, lane), accI[c]);
        accJ[c] = wmma16(a, load_bt_frag(wt_xj, HDIM, nb0 + c * 16, kc * 32, lane), accJ[c]);
      }
    }
#pragma unroll
    for (int c = 0; c < 4; ++c) {
      const int nn = nb0 + c * 16 + nn_c;
      const int rb = mt0 + rb_c;
#pragma unroll
      for (int v = 0; v < 8; ++v) {
        xi_sh[(rb + v) * HDIM + nn] = accI[c][v];
        xj_sh[(rb + v) * HDIM + nn] = accJ[c][v];
      }
    }
  }
  __syncthreads();

  // P3: pairwise h=relu(x_i+x_j+g), masked sum over j (algebraically pre-reduced before GEMM)
  const int h0 = tid & 127;
  const int jg = tid >> 7;
  const float wg0 = w_r1[(2 * HDIM + 0) * HDIM + h0];
  const float wg1 = w_r1[(2 * HDIM + 1) * HDIM + h0];
  const float wg2 = w_r1[(2 * HDIM + 2) * HDIM + h0];
  const float br1 = b_r1[h0];

  for (int i = 0; i < KAG; ++i) {
    float4* buf = geo_sh[i & 1];
    if (tid < KAG) {
      const int j = tid;
      const float dx = s_sh[j * FDIM + 0] - s_sh[i * FDIM + 0];
      const float dy = s_sh[j * FDIM + 1] - s_sh[i * FDIM + 1];
      const float dvx = s_sh[j * FDIM + 2] - s_sh[i * FDIM + 2];
      const float dvy = s_sh[j * FDIM + 3] - s_sh[i * FDIM + 3];
      const float sq = dx * dx + dy * dy;
      const float dist = (sq == 0.f) ? 0.f : sqrtf(sq);
      const float sx = ((dx == 0.f) && (dy == 0.f)) ? 1.f : dx;
      const float ang = atan2f(dy, sx);
      const float sv = dvx * dvx + dvy * dvy;
      const float sdf = (sv == 0.f) ? 0.f : sqrtf(sv);
      buf[j] = make_float4(dist, ang, sdf, m_sh[j]);
    }
    __syncthreads();
    const float base = xi_sh[i * HDIM + h0] + br1;
    float acc = 0.f;
    for (int j = jg; j < KAG; j += 2) {
      const float4 g4 = buf[j];
      const float t = xj_sh[j * HDIM + h0] + g4.x * wg0 + g4.y * wg1 + g4.z * wg2 + base;
      acc += g4.w * fmaxf(t, 0.f);
    }
    atomicAdd(&hsum_sh[i * HDIM + h0], m_sh[i] * acc);
  }
  __syncthreads();

  // P4: hn = hsum/denom -> f16 (reuse xj region), pad rows zeroed
  _Float16* hn_h = reinterpret_cast<_Float16*>(xj_sh);
  {
    const float S = S_sh;
    for (int idx = tid; idx < 64 * HDIM; idx += 256) {
      const int i = idx >> 7;
      float v = 0.f;
      if (i < KAG) {
        const float den = fmaxf(m_sh[i] * S, 1.f);
        v = hsum_sh[idx] / den;
      }
      hn_h[idx] = (_Float16)v;
    }
  }
  __syncthreads();

  // P5: agg = hn @ w_r2 + (c/den)*b_r2 -> f16 (reuse hsum region)
  _Float16* agg_h = reinterpret_cast<_Float16*>(hsum_sh);
  {
    const float S = S_sh;
    v8f acc[4] = {};
#pragma unroll
    for (int kc = 0; kc < 4; ++kc) {
      const v16h a = load_a_frag(hn_h, HDIM, mt0, kc * 32, lane);
#pragma unroll
      for (int c = 0; c < 4; ++c)
        acc[c] = wmma16(a, load_bt_frag(wt_r2, HDIM, nb0 + c * 16, kc * 32, lane), acc[c]);
    }
#pragma unroll
    for (int c = 0; c < 4; ++c) {
      const int nn = nb0 + c * 16 + nn_c;
      const float brn = b_r2[nn];
      const int rb = mt0 + rb_c;
#pragma unroll
      for (int v = 0; v < 8; ++v) {
        const int m = rb + v;
        const float cm = m_sh[m] * S;              // 0 on pad rows
        const float den = fmaxf(cm, 1.f);
        agg_h[m * HDIM + nn] = (_Float16)(acc[c][v] + (cm / den) * brn);
      }
    }
  }
  __syncthreads();

  // P6: combined = relu(af@w_g1 + agg@w_g2 + b_g); masked pool (1 atomic per lane per tile)
  {
    v8f acc[4] = {};
#pragma unroll
    for (int kc = 0; kc < 4; ++kc) {
      const v16h a = load_a_frag(af_h, HDIM, mt0, kc * 32, lane);
#pragma unroll
      for (int c = 0; c < 4; ++c)
        acc[c] = wmma16(a, load_bt_frag(wt_g1, HDIM, nb0 + c * 16, kc * 32, lane), acc[c]);
    }
#pragma unroll
    for (int kc = 0; kc < 4; ++kc) {
      const v16h a = load_a_frag(agg_h, HDIM, mt0, kc * 32, lane);
#pragma unroll
      for (int c = 0; c < 4; ++c)
        acc[c] = wmma16(a, load_bt_frag(wt_g2, HDIM, nb0 + c * 16, kc * 32, lane), acc[c]);
    }
#pragma unroll
    for (int c = 0; c < 4; ++c) {
      const int nn = nb0 + c * 16 + nn_c;
      const float bg = b_g[nn];
      const int rb = mt0 + rb_c;
      float part = 0.f;
#pragma unroll
      for (int v = 0; v < 8; ++v)
        part += m_sh[rb + v] * fmaxf(acc[c][v] + bg, 0.f);   // pad rows weight 0
      atomicAdd(&pooled_sh[nn], part);
    }
  }
  __syncthreads();

  // P7: pooled (masked mean over agents) -> f16 for final WMMA MLP
  if (tid < HDIM) {
    const float den = fmaxf(S_sh, 1.f);
    pooled_out[sc * HDIM + tid] = (_Float16)(pooled_sh[tid] / den);
  }
}

// ---- kernels 2/3: tiled WMMA GEMM, block = 64x64, wave = 16-row band x two 16-col tiles ----
template <int KDIM, bool RELU, bool OUTF16>
__global__ void __launch_bounds__(256)
gemm_kernel(const _Float16* __restrict__ A, const _Float16* __restrict__ Bt,
            const float* __restrict__ bias, float* __restrict__ outF,
            _Float16* __restrict__ outH, int N) {
  const int lane = threadIdx.x & 31;
  const int wave = threadIdx.x >> 5;
  const int m0 = blockIdx.x * 64 + ((wave >> 1) << 4);
  const int nb = blockIdx.y * 64 + ((wave & 1) << 5);   // two adjacent n-tiles
  v8f acc[2] = {};
#pragma unroll
  for (int kc = 0; kc < KDIM / 32; ++kc) {
    const v16h a = load_a_frag(A, KDIM, m0, kc * 32, lane);
#pragma unroll
    for (int c = 0; c < 2; ++c)
      acc[c] = wmma16(a, load_bt_frag(Bt, KDIM, nb + c * 16, kc * 32, lane), acc[c]);
  }
#pragma unroll
  for (int c = 0; c < 2; ++c) {
    const int nn = nb + c * 16 + (lane & 15);
    const int rb = m0 + ((lane >> 4) << 3);
    const float bv = bias[nn];
#pragma unroll
    for (int v = 0; v < 8; ++v) {
      float val = acc[c][v] + bv;
      if (RELU) val = fmaxf(val, 0.f);
      if (OUTF16) outH[(rb + v) * N + nn] = (_Float16)val;
      else        outF[(rb + v) * N + nn] = val;
    }
  }
}

extern "C" void kernel_launch(void* const* d_in, const int* in_sizes, int n_in,
                              void* d_out, int out_size, void* d_ws, size_t ws_size,
                              hipStream_t stream) {
  (void)in_sizes; (void)n_in; (void)out_size; (void)ws_size;
  const float* states = (const float*)d_in[0];
  const float* masks  = (const float*)d_in[1];
  const float* w_a  = (const float*)d_in[2];
  const float* b_a  = (const float*)d_in[3];
  const float* w_r1 = (const float*)d_in[4];
  const float* b_r1 = (const float*)d_in[5];
  const float* w_r2 = (const float*)d_in[6];
  const float* b_r2 = (const float*)d_in[7];
  const float* w_g  = (const float*)d_in[8];
  const float* b_g  = (const float*)d_in[9];
  const float* w_f1 = (const float*)d_in[10];
  const float* b_f1 = (const float*)d_in[11];
  const float* w_f2 = (const float*)d_in[12];
  const float* b_f2 = (const float*)d_in[13];

  _Float16* ws = (_Float16*)d_ws;
  _Float16* wt_xi = ws;
  _Float16* wt_xj = wt_xi + 128 * 128;
  _Float16* wt_r2 = wt_xj + 128 * 128;
  _Float16* wt_g1 = wt_r2 + 128 * 128;
  _Float16* wt_g2 = wt_g1 + 128 * 128;
  _Float16* wt_f1 = wt_g2 + 128 * 128;        // [256n][128k]
  _Float16* wt_f2 = wt_f1 + 256 * 128;        // [256n][256k]
  _Float16* pooledh = wt_f2 + 256 * 256;      // [256m][128k]
  _Float16* lat1h   = pooledh + 256 * 128;    // [256m][256k]

  auto cvt = [&](const float* src, _Float16* dst, int K, int N) {
    const int tot = K * N;
    convert_t_kernel<<<(tot + 255) / 256, 256, 0, stream>>>(src, dst, K, N);
  };
  cvt(w_r1,             wt_xi, 128, 128);
  cvt(w_r1 + 128 * 128, wt_xj, 128, 128);
  cvt(w_r2,             wt_r2, 128, 128);
  cvt(w_g,              wt_g1, 128, 128);
  cvt(w_g + 128 * 128,  wt_g2, 128, 128);
  cvt(w_f1,             wt_f1, 128, 256);
  cvt(w_f2,             wt_f2, 256, 256);

  scene_kernel<<<NSC, 256, 0, stream>>>(states, masks, w_a, b_a, w_r1, b_r1, b_r2, b_g,
                                        wt_xi, wt_xj, wt_r2, wt_g1, wt_g2, pooledh);

  gemm_kernel<128, true,  true ><<<dim3(4, 4), 256, 0, stream>>>(pooledh, wt_f1, b_f1, nullptr, lat1h, 256);
  gemm_kernel<256, false, false><<<dim3(4, 4), 256, 0, stream>>>(lat1h,   wt_f2, b_f2, (float*)d_out, nullptr, 256);
}
